// MultiHeadsAtten_35631048688254
// MI455X (gfx1250) — compile-verified, matrix operand order checked by reading, not run on hardware
//
#include <hip/hip_runtime.h>
#include <hip/hip_bf16.h>

typedef __bf16 bf16_t;
typedef __attribute__((ext_vector_type(16))) __bf16 v16bf;
typedef __attribute__((ext_vector_type(8)))  float  v8f;

#define DD 1024
#define HH 16
#define DH 64
#define SS 2048
#define BB 4

static __device__ __forceinline__ v8f wmma_bf16(v16bf a, v16bf b, v8f c) {
  return __builtin_amdgcn_wmma_f32_16x16x32_bf16(false, a, false, b, (short)0, c,
                                                 false, false);
}

// Async global->LDS copy, 16B per lane, tracked by ASYNCcnt (CDNA5).
// lds_off = LDS byte offset (low 32 bits of a generic shared pointer).
static __device__ __forceinline__ void async_b128(unsigned int lds_off,
                                                  const bf16_t* g) {
  asm volatile("global_load_async_to_lds_b128 %0, %1, off"
               :
               : "v"(lds_off), "v"(g)
               : "memory");
}
#define LDS_OFF(p) ((unsigned int)(unsigned long long)(p))
#define WAIT_ASYNC(n) asm volatile("s_wait_asynccnt " #n ::: "memory")

// A-matrix fragment (16x32 bf16): element e of lane L holds
// row M = (L&15), K = koff + e + (e&8) + 8*(L>=16). koff carries khalf+chunk.
static __device__ __forceinline__ v16bf frag_a(const bf16_t* base, int ld, int row,
                                               int koff) {
  v16bf r;
#pragma unroll
  for (int e = 0; e < 16; e += 2) {
    int k = koff + e + (e & 8);
    r[e]     = base[row * ld + k];
    r[e + 1] = base[row * ld + k + 1];
  }
  return r;
}

// B-matrix fragment (32x16 bf16) from a [K x N] row-major tile: base[k*ld + col]
static __device__ __forceinline__ v16bf frag_b(const bf16_t* base, int ld, int col,
                                               int koff) {
  v16bf r;
#pragma unroll
  for (int e = 0; e < 16; e += 2) {
    int k = koff + e + (e & 8);
    r[e]     = base[k * ld + col];
    r[e + 1] = base[(k + 1) * ld + col];
  }
  return r;
}

// B-matrix fragment gathered from a row-major [N x K] source (i.e. K^T access)
static __device__ __forceinline__ v16bf frag_bt(const bf16_t* base, int ld, int col,
                                                int koff) {
  v16bf r;
#pragma unroll
  for (int e = 0; e < 16; e += 2) {
    int k = koff + e + (e & 8);
    r[e]     = base[col * ld + k];
    r[e + 1] = base[col * ld + k + 1];
  }
  return r;
}

// ---------------- LayerNorm + cast to bf16 (one row of D=1024 per block) -------
__global__ __launch_bounds__(256) void ln_cast_kernel(const float* __restrict__ x,
                                                      const float* __restrict__ gamma,
                                                      const float* __restrict__ beta,
                                                      bf16_t* __restrict__ y) {
  __shared__ float ssum[256];
  __shared__ float ssq[256];
  const int tid = threadIdx.x;
  const size_t row = blockIdx.x;
  const float* xr = x + row * DD;
  float4 vv = *(const float4*)(xr + tid * 4);
  ssum[tid] = vv.x + vv.y + vv.z + vv.w;
  ssq[tid] = vv.x * vv.x + vv.y * vv.y + vv.z * vv.z + vv.w * vv.w;
  __syncthreads();
#pragma unroll
  for (int off = 128; off; off >>= 1) {
    if (tid < off) {
      ssum[tid] += ssum[tid + off];
      ssq[tid] += ssq[tid + off];
    }
    __syncthreads();
  }
  float mu = ssum[0] * (1.0f / DD);
  float var = ssq[0] * (1.0f / DD) - mu * mu;
  float rs = rsqrtf(var + 1e-6f);
  bf16_t* yr = y + row * DD;
#pragma unroll
  for (int i = 0; i < 4; ++i) {
    int c = tid * 4 + i;
    float val = ((&vv.x)[i] - mu) * rs * gamma[c] + beta[c];
    yr[c] = (bf16_t)val;
  }
}

// ---------------- fp32 -> bf16 weight convert ---------------------------------
__global__ __launch_bounds__(256) void f2bf_kernel(const float* __restrict__ x,
                                                   bf16_t* __restrict__ y, int n) {
  int i = (blockIdx.x * 256 + threadIdx.x) * 4;
  if (i < n) {
    float4 vv = *(const float4*)(x + i);
    y[i] = (bf16_t)vv.x;
    y[i + 1] = (bf16_t)vv.y;
    y[i + 2] = (bf16_t)vv.z;
    y[i + 3] = (bf16_t)vv.w;
  }
}

// ---------------- bf16 GEMM with bias (+optional fp32 residual) ---------------
// C[M,N] = A[M,K] * Bw[K,N] + bias (+ resid). 128x128 block tile, 8 waves,
// each wave a 32x64 sub-tile = 2x4 grid of v_wmma_f32_16x16x32_bf16.
// Double-buffered LDS tiles filled by GLOBAL_LOAD_ASYNC_TO_LDS_B128 so
// global->LDS traffic for tile t+1 overlaps the WMMAs of tile t.
template <bool OUT_BF16>
__global__ __launch_bounds__(256) void gemm_bias_kernel(
    const bf16_t* __restrict__ A, const bf16_t* __restrict__ Bw,
    const float* __restrict__ bias, const float* __restrict__ resid,
    void* __restrict__ outp, int M, int N, int K) {
  __shared__ __align__(16) bf16_t As[2][128][40];   // padded: conflict-free frags
  __shared__ __align__(16) bf16_t Bs[2][32][136];
  const int tid = threadIdx.x;
  const int lane = tid & 31;
  const int wid = tid >> 5;
  const int bm = blockIdx.y * 128;
  const int bn = blockIdx.x * 128;
  const int wm = (wid >> 1) * 32;
  const int wn = (wid & 1) * 64;
  const int lr = lane & 15;
  const int khalf = (lane >> 4) * 8;

  // per-thread async-copy assignment (16 bf16 = 2 x b128 each for A and B)
  const int ar = tid >> 1, ac = (tid & 1) * 16;
  const int br = tid >> 3, bc = (tid & 7) * 16;
  const bf16_t* aptr = A + (size_t)(bm + ar) * K + ac;
  const bf16_t* bptr = Bw + (size_t)br * N + bn + bc;

  v8f acc[2][4];
#pragma unroll
  for (int i = 0; i < 2; ++i)
#pragma unroll
    for (int j = 0; j < 4; ++j) acc[i][j] = (v8f){0, 0, 0, 0, 0, 0, 0, 0};

  // prologue: tile 0 -> buffer 0 (4 async ops per wave)
  async_b128(LDS_OFF(&As[0][ar][ac]), aptr);
  async_b128(LDS_OFF(&As[0][ar][ac + 8]), aptr + 8);
  async_b128(LDS_OFF(&Bs[0][br][bc]), bptr);
  async_b128(LDS_OFF(&Bs[0][br][bc + 8]), bptr + 8);

  const int T = K / 32;
  int buf = 0;
  for (int t = 0; t < T; ++t) {
    if (t + 1 < T) {
      const bf16_t* an = aptr + (t + 1) * 32;
      const bf16_t* bn2 = bptr + (size_t)(t + 1) * 32 * N;
      async_b128(LDS_OFF(&As[buf ^ 1][ar][ac]), an);
      async_b128(LDS_OFF(&As[buf ^ 1][ar][ac + 8]), an + 8);
      async_b128(LDS_OFF(&Bs[buf ^ 1][br][bc]), bn2);
      async_b128(LDS_OFF(&Bs[buf ^ 1][br][bc + 8]), bn2 + 8);
      WAIT_ASYNC(0x4);  // tile t complete (async ops retire in order)
    } else {
      WAIT_ASYNC(0x0);
    }
    __syncthreads();  // publish all waves' LDS writes for tile t

    v16bf af[2], bfr[4];
#pragma unroll
    for (int mt = 0; mt < 2; ++mt)
      af[mt] = frag_a(&As[buf][0][0], 40, wm + 16 * mt + lr, khalf);
#pragma unroll
    for (int nt = 0; nt < 4; ++nt)
      bfr[nt] = frag_b(&Bs[buf][0][0], 136, wn + 16 * nt + lr, khalf);
#pragma unroll
    for (int mt = 0; mt < 2; ++mt)
#pragma unroll
      for (int nt = 0; nt < 4; ++nt)
        acc[mt][nt] = wmma_bf16(af[mt], bfr[nt], acc[mt][nt]);

    __syncthreads();  // everyone done reading buf before it is refilled
    buf ^= 1;
  }

#pragma unroll
  for (int mt = 0; mt < 2; ++mt) {
#pragma unroll
    for (int nt = 0; nt < 4; ++nt) {
#pragma unroll
      for (int r = 0; r < 8; ++r) {
        int row = bm + wm + 16 * mt + r + khalf;  // C layout: M = reg + 8*(lane>=16)
        int col = bn + wn + 16 * nt + lr;
        float vv = acc[mt][nt][r] + bias[col];
        size_t idx = (size_t)row * N + col;
        if (resid) vv += resid[idx];
        if (OUT_BF16)
          ((bf16_t*)outp)[idx] = (bf16_t)vv;
        else
          ((float*)outp)[idx] = vv;
      }
    }
  }
}

// ---------------- causal flash attention (bf16 in, bf16 out) ------------------
// grid = (S/128, H, B); 8 waves/block; wave owns 16 query rows; 32-key chunks.
__global__ __launch_bounds__(256) void attention_kernel(
    const bf16_t* __restrict__ q, const bf16_t* __restrict__ k,
    const bf16_t* __restrict__ v, bf16_t* __restrict__ o) {
  __shared__ bf16_t pbuf[8][16][40];  // per-wave P staging (C-layout -> A-layout)
  const int tid = threadIdx.x;
  const int lane = tid & 31;
  const int wid = tid >> 5;
  const int lr = lane & 15;
  const int khalf = (lane >> 4) * 8;
  const int b = blockIdx.z;
  const int h = blockIdx.y;
  const int qbase = blockIdx.x * 128 + wid * 16;
  const size_t hoff = (size_t)h * DH;

  const bf16_t* qbp = q + ((size_t)(b * SS + qbase)) * DD + hoff;
  v16bf qa[2];
#pragma unroll
  for (int c = 0; c < 2; ++c) qa[c] = frag_a(qbp, DD, lr, 32 * c + khalf);

  v8f oacc[4];
#pragma unroll
  for (int i = 0; i < 4; ++i) oacc[i] = (v8f){0, 0, 0, 0, 0, 0, 0, 0};
  float mrun[8], lrun[8];
#pragma unroll
  for (int r = 0; r < 8; ++r) {
    mrun[r] = -3.0e38f;
    lrun[r] = 0.0f;
  }

  for (int j0 = 0; j0 < qbase + 16; j0 += 32) {
    v8f s0 = (v8f){0, 0, 0, 0, 0, 0, 0, 0};
    v8f s1 = (v8f){0, 0, 0, 0, 0, 0, 0, 0};
    const bf16_t* kbp0 = k + ((size_t)(b * SS + j0)) * DD + hoff;
    const bf16_t* kbp1 = kbp0 + (size_t)16 * DD;
#pragma unroll
    for (int c = 0; c < 2; ++c) {
      v16bf kb0 = frag_bt(kbp0, DD, lr, 32 * c + khalf);
      s0 = wmma_bf16(qa[c], kb0, s0);
      v16bf kb1 = frag_bt(kbp1, DD, lr, 32 * c + khalf);
      s1 = wmma_bf16(qa[c], kb1, s1);
    }
    // causal mask + online softmax; row r's 16 cols live across a 16-lane half
    float p0[8], p1[8];
#pragma unroll
    for (int r = 0; r < 8; ++r) {
      int row = qbase + r + khalf;
      float a0 = ((j0 + lr) <= row) ? s0[r] : -3.0e38f;
      float a1 = ((j0 + 16 + lr) <= row) ? s1[r] : -3.0e38f;
      float rm = fmaxf(a0, a1);
#pragma unroll
      for (int off = 8; off; off >>= 1) rm = fmaxf(rm, __shfl_xor(rm, off, 32));
      float mnew = fmaxf(mrun[r], rm);
      float scale = __expf(mrun[r] - mnew);
      float e0 = __expf(a0 - mnew);
      float e1 = __expf(a1 - mnew);
      float rs = e0 + e1;
#pragma unroll
      for (int off = 8; off; off >>= 1) rs += __shfl_xor(rs, off, 32);
      lrun[r] = lrun[r] * scale + rs;
      mrun[r] = mnew;
#pragma unroll
      for (int nt = 0; nt < 4; ++nt) oacc[nt][r] *= scale;
      p0[r] = e0;
      p1[r] = e1;
    }
    // transpose P through per-wave LDS (C-layout -> A-layout); same-wave DS ops
    // are in-order, explicit s_wait_dscnt covers the store->load dependency
#pragma unroll
    for (int r = 0; r < 8; ++r) {
      int m = r + khalf;
      pbuf[wid][m][lr] = (bf16_t)p0[r];
      pbuf[wid][m][lr + 16] = (bf16_t)p1[r];
    }
    asm volatile("s_wait_dscnt 0x0" ::: "memory");
    v16bf pf = frag_a(&pbuf[wid][0][0], 40, lr, khalf);
    const bf16_t* vbp = v + ((size_t)(b * SS + j0)) * DD + hoff;
#pragma unroll
    for (int nt = 0; nt < 4; ++nt) {
      v16bf vf = frag_b(vbp + nt * 16, DD, lr, khalf);
      oacc[nt] = wmma_bf16(pf, vf, oacc[nt]);
    }
    asm volatile("" ::: "memory");
  }

  bf16_t* obp = o + ((size_t)(b * SS + qbase)) * DD + hoff;
#pragma unroll
  for (int nt = 0; nt < 4; ++nt) {
#pragma unroll
    for (int r = 0; r < 8; ++r) {
      int m = r + khalf;
      float val = oacc[nt][r] / lrun[r];
      obp[(size_t)m * DD + nt * 16 + lr] = (bf16_t)val;
    }
  }
}

// ------------------------------------------------------------------------------
extern "C" void kernel_launch(void* const* d_in, const int* in_sizes, int n_in,
                              void* d_out, int out_size, void* d_ws, size_t ws_size,
                              hipStream_t stream) {
  const float* Q = (const float*)d_in[0];
  const float* K = (const float*)d_in[1];
  const float* V = (const float*)d_in[2];
  const float* Wq = (const float*)d_in[3];
  const float* bq = (const float*)d_in[4];
  const float* Wk = (const float*)d_in[5];
  const float* bk = (const float*)d_in[6];
  const float* Wv = (const float*)d_in[7];
  const float* bv = (const float*)d_in[8];
  const float* Wo = (const float*)d_in[9];
  const float* bo = (const float*)d_in[10];
  const float* gamma = (const float*)d_in[11];
  const float* beta = (const float*)d_in[12];
  float* out = (float*)d_out;

  char* ws = (char*)d_ws;
  const size_t SZA = (size_t)BB * SS * DD * sizeof(bf16_t);  // 16 MB
  const size_t SZW = (size_t)DD * DD * sizeof(bf16_t);       // 2 MB
  bf16_t* lnQ = (bf16_t*)(ws);
  bf16_t* lnK = (bf16_t*)(ws + SZA);
  bf16_t* lnV = (bf16_t*)(ws + 2 * SZA);
  bf16_t* qb = (bf16_t*)(ws + 3 * SZA);
  bf16_t* kb = (bf16_t*)(ws + 4 * SZA);
  bf16_t* vb = (bf16_t*)(ws + 5 * SZA);
  bf16_t* ao = (bf16_t*)(ws + 6 * SZA);
  bf16_t* WqB = (bf16_t*)(ws + 7 * SZA);
  bf16_t* WkB = (bf16_t*)(ws + 7 * SZA + SZW);
  bf16_t* WvB = (bf16_t*)(ws + 7 * SZA + 2 * SZW);
  bf16_t* WoB = (bf16_t*)(ws + 7 * SZA + 3 * SZW);

  const int M = BB * SS;  // 8192
  dim3 blk(256);

  ln_cast_kernel<<<M, blk, 0, stream>>>(Q, gamma, beta, lnQ);
  ln_cast_kernel<<<M, blk, 0, stream>>>(K, gamma, beta, lnK);
  ln_cast_kernel<<<M, blk, 0, stream>>>(V, gamma, beta, lnV);

  f2bf_kernel<<<1024, blk, 0, stream>>>(Wq, WqB, DD * DD);
  f2bf_kernel<<<1024, blk, 0, stream>>>(Wk, WkB, DD * DD);
  f2bf_kernel<<<1024, blk, 0, stream>>>(Wv, WvB, DD * DD);
  f2bf_kernel<<<1024, blk, 0, stream>>>(Wo, WoB, DD * DD);

  dim3 ggrid(DD / 128, M / 128);
  gemm_bias_kernel<true><<<ggrid, blk, 0, stream>>>(lnQ, WqB, bq, nullptr, qb, M, DD, DD);
  gemm_bias_kernel<true><<<ggrid, blk, 0, stream>>>(lnK, WkB, bk, nullptr, kb, M, DD, DD);
  gemm_bias_kernel<true><<<ggrid, blk, 0, stream>>>(lnV, WvB, bv, nullptr, vb, M, DD, DD);

  dim3 agrid(SS / 128, HH, BB);
  attention_kernel<<<agrid, blk, 0, stream>>>(qb, kb, vb, ao);

  gemm_bias_kernel<false><<<ggrid, blk, 0, stream>>>(ao, WoB, bo, Q, out, M, DD, DD);
}
